// Model_22797686407311
// MI455X (gfx1250) — matrix operand order repaired and checked by reading, not compiled, not run on hardware
//
#include <hip/hip_runtime.h>
#include <math.h>

// ---------------- types ----------------
typedef __bf16 bf16;
typedef bf16  v16bf __attribute__((ext_vector_type(16)));
typedef bf16  v8bf  __attribute__((ext_vector_type(8)));
typedef float v8f   __attribute__((ext_vector_type(8)));
typedef unsigned int u32x4 __attribute__((ext_vector_type(4)));
typedef int          i32x4 __attribute__((ext_vector_type(4)));
typedef int          i32x8 __attribute__((ext_vector_type(8)));

#define SEQN  2048
#define BATCH 64
#define HD    512

union AFrag { v16bf v; v8bf h[2]; };

__device__ __forceinline__ unsigned lds_addr_of(const void* p) {
  // generic LDS pointers carry the byte offset in the low 32 bits
  return (unsigned)(unsigned long long)(uintptr_t)p;
}

// Issue a 2-D Tensor-Data-Mover load (bf16 elements) into LDS.  D# layout per
// CDNA5 ISA ch.8: group0 = {count, lds_addr, global_addr, type=2}, group1 packs
// data_size=2B, tensor dims, tile dims, dim0 stride.  Tracked on TENSORcnt.
__device__ __forceinline__ void tdm_load_2d(const void* gptr, unsigned ldsAddr,
                                            unsigned dim0, unsigned dim1,
                                            unsigned tile0, unsigned tile1,
                                            unsigned stride) {
  unsigned long long ga = (unsigned long long)(uintptr_t)gptr;
  u32x4 g0 = { 1u,                                   // count=1, user descriptor
               ldsAddr,                               // lds_addr (bytes)
               (unsigned)ga,                          // global_addr[31:0]
               (unsigned)((ga >> 32) & 0x1FFFFFFull) | (2u << 30) };
  i32x8 g1 = { (int)(1u << 16),                       // workgroup_mask=0, data_size=1 (2B)
               (int)((dim0 & 0xFFFFu) << 16),         // tensor_dim0 lo16
               (int)((dim0 >> 16) | ((dim1 & 0xFFFFu) << 16)),
               (int)(((dim1 >> 16) & 0xFFFFu) | (tile0 << 16)),
               (int)(tile1 & 0xFFFFu),                // tile_dim1 | tile_dim2=0
               (int)stride,                           // tensor_dim0_stride lo32
               0, 0 };
  i32x4 z4 = { 0, 0, 0, 0 };
  i32x8 z8 = { 0, 0, 0, 0, 0, 0, 0, 0 };
  __builtin_amdgcn_tensor_load_to_lds(g0, g1, z4, z4, z8, 0);
}

// ---------------- prep: f32 -> bf16 ----------------
__global__ void cvt_f32_bf16(const float* __restrict__ src, bf16* __restrict__ dst, int n) {
  for (int i = blockIdx.x * blockDim.x + threadIdx.x; i < n; i += gridDim.x * blockDim.x)
    dst[i] = (bf16)src[i];
}

__global__ void bias_sum(const float* __restrict__ a, const float* __restrict__ b,
                         float* __restrict__ o) {
  int i = threadIdx.x;
  o[i] = a[i] + b[i];
}

// ---------------- GEMM: out[M,512] = A[M,512] @ W[512,512]^T + bias ----------------
// Block = 1024 threads = 32 waves: 8 M-tiles x 4 N-groups(128 cols); 128 rows/block.
// W K-chunks (512x32 bf16 = 32 KB) stream through a TDM ping-pong pair so the DMA
// for chunk j+1 overlaps the 8 WMMAs of chunk j.  One barrier per chunk.
__global__ __launch_bounds__(1024, 1)
void gemm_xw(const bf16* __restrict__ A, const bf16* __restrict__ W,
             const float* __restrict__ bias, bf16* __restrict__ out) {
  __shared__ bf16 lds_b[2][HD * 32];                // 2 x 32 KB ping-pong
  const int tid   = threadIdx.x;
  const int lane  = tid & 31;
  const int wave  = tid >> 5;
  const int lx    = lane & 15;
  const int khalf = lane >> 4;
  const int mtile  = wave & 7;
  const int ngroup = wave >> 3;
  const int rowbase = blockIdx.x * 128 + mtile * 16;

  const unsigned ldsB0 = lds_addr_of(&lds_b[0][0]);
  const unsigned ldsB1 = lds_addr_of(&lds_b[1][0]);

  if (wave == 0)                                    // prime the pipe: chunk 0
    tdm_load_2d(W, ldsB0, HD, HD, 32, HD, HD);

  v8f acc[8] = {};

  for (int j = 0; j < 16; ++j) {
    if (wave == 0) __builtin_amdgcn_s_wait_tensorcnt(0);
    __syncthreads();                                // chunk j visible to all waves
    if (wave == 0 && j < 15)                        // overlap DMA(j+1) with WMMA(j)
      tdm_load_2d((const bf16*)W + (j + 1) * 32, ((j + 1) & 1) ? ldsB1 : ldsB0,
                  HD, HD, 32, HD, HD);

    const bf16* bcur = &lds_b[j & 1][0];
    const bf16* arow = A + (size_t)(rowbase + lx) * HD + j * 32 + khalf * 16;
    AFrag a;
    a.h[0] = *(const v8bf*)(arow);
    a.h[1] = *(const v8bf*)(arow + 8);
    if (j < 14) __builtin_prefetch(arow + 64, 0, 3);   // A stream, 2 chunks ahead

#pragma unroll
    for (int nt = 0; nt < 8; ++nt) {
      int coln = ngroup * 128 + nt * 16 + lx;
      v16bf b = *(const v16bf*)&bcur[coln * 32 + khalf * 16];
      acc[nt] = __builtin_amdgcn_wmma_f32_16x16x32_bf16(
          false, a.v, false, b, (short)0, acc[nt], false, false);
    }
  }

#pragma unroll
  for (int nt = 0; nt < 8; ++nt) {
    int coln = ngroup * 128 + nt * 16 + lx;
    float bv = bias[coln];
#pragma unroll
    for (int r = 0; r < 8; ++r) {
      int row = rowbase + khalf * 8 + r;            // C layout: lanes16-31 -> M+8
      out[(size_t)row * HD + coln] = (bf16)(acc[nt][r] + bv);
    }
  }
}

// ---------------- recurrent scan (one persistent WGP workgroup) ----------------
// h_t = tanh(xw_t + h_{t-1} @ W_hh^T).  W_hh lives in VGPRs (16 KB/wave = 128
// VGPR/lane), h in LDS (64 KB), and xw_t is TDM-prefetched one step ahead into a
// 2 x 64 KB LDS ping-pong (192 KB LDS total) so the serial loop never waits on
// strided HBM loads.  Wave w owns output columns [16w, 16w+16).
__global__ __launch_bounds__(1024, 1)
void rnn_scan(const bf16* __restrict__ xw,       // [SEQ, BATCH, HD] bf16 (bias folded in)
              const bf16* __restrict__ Whh,      // [HD, HD] row-major
              bf16* __restrict__ y16,            // layer-0 output (bf16) or null
              float* __restrict__ y32) {         // layer-1 output (f32) or null
  __shared__ bf16 h_lds[BATCH * HD];             // 64 KB
  __shared__ bf16 xw_lds[2][BATCH * HD];         // 128 KB ping-pong
  const int tid   = threadIdx.x;
  const int lane  = tid & 31;
  const int wave  = tid >> 5;
  const int lx    = lane & 15;
  const int khalf = lane >> 4;
  const int col   = wave * 16 + lx;

  const unsigned ldsX0 = lds_addr_of(&xw_lds[0][0]);
  const unsigned ldsX1 = lds_addr_of(&xw_lds[1][0]);
  const unsigned XWTILE = BATCH * HD;            // 32768 elements = 64 KB

  // This wave's B slice of W_hh into registers: 16 chunks of 32x16 bf16.
  v16bf bfrag[16];
#pragma unroll
  for (int j = 0; j < 16; ++j)
    bfrag[j] = *(const v16bf*)&Whh[(size_t)col * HD + j * 32 + khalf * 16];

  if (wave == 0)                                  // prime: xw_0 -> buffer 0
    tdm_load_2d(xw, ldsX0, XWTILE, 1, XWTILE, 1, XWTILE);

  for (int i = tid; i < BATCH * HD; i += 1024) h_lds[i] = (bf16)0.0f;  // h0 = 0
  __syncthreads();

  for (int t = 0; t < SEQN; ++t) {
    if (wave == 0) __builtin_amdgcn_s_wait_tensorcnt(0);
    __syncthreads();                              // xw_t staged + h_{t-1} published
    if (wave == 0 && t + 1 < SEQN)                // DMA xw_{t+1} under this step's WMMAs
      tdm_load_2d(xw + (size_t)(t + 1) * XWTILE, ((t + 1) & 1) ? ldsX1 : ldsX0,
                  XWTILE, 1, XWTILE, 1, XWTILE);

    v8f acc[4] = {};
#pragma unroll
    for (int j = 0; j < 16; ++j) {
#pragma unroll
      for (int m = 0; m < 4; ++m) {               // 4 M-tiles cover BATCH=64
        const bf16* hrow = &h_lds[(m * 16 + lx) * HD + j * 32 + khalf * 16];
        AFrag a;
        a.h[0] = *(const v8bf*)(hrow);
        a.h[1] = *(const v8bf*)(hrow + 8);
        acc[m] = __builtin_amdgcn_wmma_f32_16x16x32_bf16(
            false, a.v, false, bfrag[j], (short)0, acc[m], false, false);
      }
    }
    __syncthreads();                              // all reads of h_{t-1} complete

    const bf16* xcur = &xw_lds[t & 1][0];
#pragma unroll
    for (int m = 0; m < 4; ++m) {
#pragma unroll
      for (int r = 0; r < 8; ++r) {
        int row = m * 16 + khalf * 8 + r;
        float v  = acc[m][r] + (float)xcur[row * HD + col];
        float hv = tanhf(v);
        h_lds[row * HD + col] = (bf16)hv;
        if (y32) y32[(size_t)t * BATCH * HD + row * HD + col] = hv;
        else     y16[(size_t)t * BATCH * HD + row * HD + col] = (bf16)hv;
      }
    }
  }
}

// ---------------- launch ----------------
extern "C" void kernel_launch(void* const* d_in, const int* in_sizes, int n_in,
                              void* d_out, int out_size, void* d_ws, size_t ws_size,
                              hipStream_t stream) {
  const float* x     = (const float*)d_in[0];
  const float* W_ih0 = (const float*)d_in[2];
  const float* W_hh0 = (const float*)d_in[3];
  const float* b_ih0 = (const float*)d_in[4];
  const float* b_hh0 = (const float*)d_in[5];
  const float* W_ih1 = (const float*)d_in[6];
  const float* W_hh1 = (const float*)d_in[7];
  const float* b_ih1 = (const float*)d_in[8];
  const float* b_hh1 = (const float*)d_in[9];

  const size_t NTOT = (size_t)SEQN * BATCH * HD;       // 67,108,864
  const int    WN   = HD * HD;                         // 262,144

  bf16*  xbuf  = (bf16*)d_ws;          // x bf16, later reused as y0 bf16
  bf16*  xw    = xbuf + NTOT;          // xW buffer (both layers)
  bf16*  wih0  = xw + NTOT;
  bf16*  whh0  = wih0 + WN;
  bf16*  wih1  = whh0 + WN;
  bf16*  whh1  = wih1 + WN;
  float* bias0 = (float*)(whh1 + WN);
  float* bias1 = bias0 + HD;

  // 1) convert inputs/weights to bf16, fuse biases
  cvt_f32_bf16<<<8192, 256, 0, stream>>>(x, xbuf, (int)NTOT);
  cvt_f32_bf16<<<1024, 256, 0, stream>>>(W_ih0, wih0, WN);
  cvt_f32_bf16<<<1024, 256, 0, stream>>>(W_hh0, whh0, WN);
  cvt_f32_bf16<<<1024, 256, 0, stream>>>(W_ih1, wih1, WN);
  cvt_f32_bf16<<<1024, 256, 0, stream>>>(W_hh1, whh1, WN);
  bias_sum<<<1, HD, 0, stream>>>(b_ih0, b_hh0, bias0);
  bias_sum<<<1, HD, 0, stream>>>(b_ih1, b_hh1, bias1);

  // 2) layer 0: hoisted input projection, then serial scan (y0 -> xbuf, bf16)
  gemm_xw<<<1024, 1024, 0, stream>>>(xbuf, wih0, bias0, xw);
  rnn_scan<<<1, 1024, 0, stream>>>(xw, whh0, xbuf, nullptr);

  // 3) layer 1: projection of y0, then serial scan writing f32 to d_out
  gemm_xw<<<1024, 1024, 0, stream>>>(xbuf, wih1, bias1, xw);
  rnn_scan<<<1, 1024, 0, stream>>>(xw, whh1, nullptr, (float*)d_out);
}
